// Siamese2_10127532884802
// MI455X (gfx1250) — matrix operand order repaired and checked by reading, not compiled, not run on hardware
//
#include <hip/hip_runtime.h>
#include <hip/hip_bf16.h>
#include <stdint.h>

typedef __attribute__((ext_vector_type(16))) __bf16 v16bf;
typedef __attribute__((ext_vector_type(8)))  __bf16 v8bf;
typedef __attribute__((ext_vector_type(4)))  __bf16 v4bf;
typedef __attribute__((ext_vector_type(8)))  float  v8f;
typedef __attribute__((ext_vector_type(4)))  unsigned int u32x4;
typedef __attribute__((ext_vector_type(8)))  int    i32x8;
typedef __attribute__((ext_vector_type(4)))  int    i32x4;

#define IMGW   2048
#define OH     256
#define NPOS1  65536        // 256*256
#define OH2    254
#define NPOS2  64516        // 254*254
#define CFEAT  64
#define KDIM1  192          // 3*8*8
#define NCLS   32
#define KDIM2  576          // 64*3*3
#define TOPK   128
#define FDIM   2048

#define ASTR1  200          // padded LDS stride for A (feat): 400B, 16B-aligned rows
#define ASTR2  584          // padded LDS stride for A (cls): 1168B, 16B-aligned rows
#define KSTR   40           // padded per-column K stride for B: 80B, 16B-aligned cols

// Tensor Data Mover availability (device pass on toolchains that ship it)
#if defined(__has_builtin)
#if __has_builtin(__builtin_amdgcn_tensor_load_to_lds)
#define HAVE_TDM 1
#endif
#endif
#ifndef HAVE_TDM
#define HAVE_TDM 0
#endif

// -------------------- workspace layout (float slots) --------------------
#define WS_X    0                               // 64*65536      x (post-relu feat)
#define WS_C    (WS_X + CFEAT*NPOS1)            // 32*64516      c (cls conv out)
#define WS_CMAX (WS_C + NCLS*NPOS2)             // 64520         channel-max map
#define WS_IDX  (WS_CMAX + 64520)               // 128 ints      top-k flat indices
#define WS_REGN (WS_IDX + 128)                  // 128*576       normalized patches
#define WS_T    (WS_REGN + TOPK*KDIM2)          // 576           summed reg vector
#define WS_ACC  (WS_T + KDIM2)                  // 2048          pre-norm descriptor

// ---- fragment loaders: two contiguous 16B LDS reads each (ds_load_b128) ----
__device__ __forceinline__ v16bf ldfragA(const __bf16* p, int hi)
{
    v8bf lo = *(const v8bf*)(p + hi * 8);
    v8bf hh = *(const v8bf*)(p + 16 + hi * 8);
    return __builtin_shufflevector(lo, hh, 0,1,2,3,4,5,6,7,8,9,10,11,12,13,14,15);
}
__device__ __forceinline__ v16bf ldfragB(const __bf16* p, int hi)
{
    v8bf lo = *(const v8bf*)(p + hi * 8);
    v8bf hh = *(const v8bf*)(p + 16 + hi * 8);
    return __builtin_shufflevector(lo, hh, 0,1,2,3,4,5,6,7,8,9,10,11,12,13,14,15);
}
__device__ __forceinline__ v8bf cvt8(float4 a, float4 b)
{
    v8bf c;
    c[0]=(__bf16)a.x; c[1]=(__bf16)a.y; c[2]=(__bf16)a.z; c[3]=(__bf16)a.w;
    c[4]=(__bf16)b.x; c[5]=(__bf16)b.y; c[6]=(__bf16)b.z; c[7]=(__bf16)b.w;
    return c;
}

#if HAVE_TDM
// TDM: load 4 consecutive image rows x 1024 floats (row stride 2048) to LDS.
// D# per CDNA5 ISA ch.8: group0 = {count/flags, lds_addr, global_addr, type=2},
// group1 = {mask/data_size, ..., tensor dims, tile 1024x4, stride0=2048}.
__device__ __forceinline__ void tdm_load_4rows(const float* gsrc, unsigned lds_off)
{
    unsigned long long ga = (unsigned long long)(uintptr_t)gsrc;
    u32x4 g0 = { 1u,                                   // count=1, user descriptor
                 lds_off,                              // lds_addr (bytes)
                 (unsigned)(ga & 0xffffffffu),         // global_addr[31:0]
                 (unsigned)(((ga >> 32) & 0x01ffffffull) | 0x80000000u) }; // ga[56:32] | type=2
    i32x8 g1 = { 0x20000,                              // data_size=2 (4B), wg_mask=0
                 0,                                    // no atomic barrier
                 0x0010,                               // tensor_dim0=1M (hi16) | td1 lo16=0
                 0x04000010,                           // tile_dim0=1024 | tensor_dim1=1M hi16
                 4,                                    // tile_dim1=4, tile_dim2=0
                 2048,                                 // tensor_dim0_stride=2048
                 0, 0 };
    i32x4 gz = { 0, 0, 0, 0 };
#if __clang_major__ >= 23
    i32x8 gz8 = { 0,0,0,0,0,0,0,0 };
    __builtin_amdgcn_tensor_load_to_lds(g0, g1, gz, gz, gz8, 0);
#else
    __builtin_amdgcn_tensor_load_to_lds(g0, g1, gz, gz, 0);
#endif
}
#endif

// repack raw f32 rows (LDS) -> column-major bf16 B panel (LDS)
__device__ __forceinline__ void repack_feat(const float* rawbuf, __bf16* dst, int tid)
{
#pragma unroll
    for (int r = 0; r < 2; ++r) {
        int u = tid + r * 256;
        int col = u & 127, g = u >> 7;
        const float4* s4 = (const float4*)(rawbuf + g * 1024 + col * 8);
        *(v8bf*)&dst[col * KSTR + g * 8] = cvt8(s4[0], s4[1]);
    }
}

// fallback staging: global f32 -> bf16 column-major panel
__device__ __forceinline__ void stage_feat(const float* __restrict__ x1, __bf16* dst,
                                           int k0, int oy, int oxb, int tid)
{
#pragma unroll
    for (int r = 0; r < 2; ++r) {
        int u = tid + r * 256;
        int col = u & 127, g = u >> 7;
        int k = k0 + g * 8;
        int c = k >> 6, ky = (k & 63) >> 3;
        const float* src = x1 + (size_t)c * (IMGW*IMGW) + (oy*8 + ky) * IMGW + (oxb + col) * 8;
        float4 fa = *(const float4*)src, fb4 = *(const float4*)(src + 4);
        *(v8bf*)&dst[col * KSTR + g * 8] = cvt8(fa, fb4);
    }
}

__device__ __forceinline__ void stage_cls(const float* __restrict__ xf, __bf16* dst,
                                          int k0, int base, int tid)
{
    for (int e = tid; e < 32 * 128; e += 256) {
        int col = e >> 5, kk = e & 31;
        int p   = base + col;
        float v = 0.f;
        if (p < NPOS2) {
            int k = k0 + kk;
            int c = k / 9, r = k - c * 9, fy = r / 3, fx = r - fy * 3;
            int oyy = p / OH2, oxx = p - oyy * OH2;
            v = xf[(size_t)c * NPOS1 + (oyy + fy) * OH + (oxx + fx)];
        }
        dst[col * KSTR + kk] = (__bf16)v;
    }
}

// =======================================================================
// Kernel 1: feat conv as GEMM  [64 x 192] x [192 x 128-position tile]
// TDM pipelined: DMA next panel while WMMAs consume current one.
// =======================================================================
__global__ __launch_bounds__(256) void k_feat_conv(
    const float* __restrict__ x1, const float* __restrict__ fw,
    const float* __restrict__ fb, float* __restrict__ xf)
{
    __shared__ __bf16 A[CFEAT * ASTR1];
    __shared__ __bf16 Bc[2][128 * KSTR];
#if HAVE_TDM
    __shared__ float raw[2][4 * 1024];
#endif

    const int tid   = threadIdx.x;
    const int lane  = tid & 31, wv = tid >> 5;
    const int lrow  = lane & 15, hi = lane >> 4;
    const int mtile = wv & 3,  nhalf = wv >> 2;
    const int base  = blockIdx.x * 128;
    const int oy = base >> 8, oxb = base & 255;

    // stage weights (64x192) bf16, float4 -> b64 stores
    for (int u = tid; u < CFEAT * KDIM1 / 4; u += 256) {
        int m = u / 48, kq = u - m * 48;
        float4 f = *(const float4*)(fw + m * KDIM1 + 4 * kq);
        v4bf c; c[0]=(__bf16)f.x; c[1]=(__bf16)f.y; c[2]=(__bf16)f.z; c[3]=(__bf16)f.w;
        *(v4bf*)&A[m * ASTR1 + 4 * kq] = c;
    }

#if HAVE_TDM
    if (wv == 0) {
        tdm_load_4rows(x1 + (size_t)0 * (IMGW*IMGW) + (oy*8) * IMGW + oxb * 8,
                       (unsigned)(uintptr_t)&raw[0][0]);
        __builtin_amdgcn_s_wait_tensorcnt(0);
    }
    __syncthreads();
    repack_feat(&raw[0][0], &Bc[0][0], tid);
    __syncthreads();
#else
    stage_feat(x1, &Bc[0][0], 0, oy, oxb, tid);
    __syncthreads();
#endif

    // preload all 6 A fragments (A is stable after first barrier)
    v16bf af[6];
#pragma unroll
    for (int ks = 0; ks < 6; ++ks)
        af[ks] = ldfragA(&A[(mtile*16 + lrow) * ASTR1 + ks * 32], hi);

    v8f zero = {0.f,0.f,0.f,0.f,0.f,0.f,0.f,0.f};
    v8f acc[4];
#pragma unroll
    for (int nt = 0; nt < 4; ++nt) acc[nt] = zero;

    for (int ks = 0; ks < 6; ++ks) {
        const int cur = ks & 1, nxt = cur ^ 1;
#if HAVE_TDM
        if (ks < 5 && wv == 0) {
            int kn = (ks + 1) * 32;
            int c = kn >> 6, ky0 = (kn & 63) >> 3;
            tdm_load_4rows(x1 + (size_t)c * (IMGW*IMGW) + (oy*8 + ky0) * IMGW + oxb * 8,
                           (unsigned)(uintptr_t)&raw[nxt][0]);
        }
#else
        if (ks < 5) stage_feat(x1, &Bc[nxt][0], (ks + 1) * 32, oy, oxb, tid);
#endif
        // batch all B fragment loads, then the WMMA chain
        {
            const __bf16* bp = &Bc[cur][(nhalf*64 + lrow) * KSTR];
            v16bf b0 = ldfragB(bp + 0 * 16 * KSTR, hi);
            v16bf b1 = ldfragB(bp + 1 * 16 * KSTR, hi);
            v16bf b2 = ldfragB(bp + 2 * 16 * KSTR, hi);
            v16bf b3 = ldfragB(bp + 3 * 16 * KSTR, hi);
            acc[0] = __builtin_amdgcn_wmma_f32_16x16x32_bf16(false, af[ks], false, b0, (short)0, acc[0], false, false);
            acc[1] = __builtin_amdgcn_wmma_f32_16x16x32_bf16(false, af[ks], false, b1, (short)0, acc[1], false, false);
            acc[2] = __builtin_amdgcn_wmma_f32_16x16x32_bf16(false, af[ks], false, b2, (short)0, acc[2], false, false);
            acc[3] = __builtin_amdgcn_wmma_f32_16x16x32_bf16(false, af[ks], false, b3, (short)0, acc[3], false, false);
        }
#if HAVE_TDM
        if (ks < 5) {
            if (wv == 0) __builtin_amdgcn_s_wait_tensorcnt(0);
            __syncthreads();
            repack_feat(&raw[nxt][0], &Bc[nxt][0], tid);
            __syncthreads();
        }
#else
        if (ks < 5) __syncthreads();
#endif
    }

    // epilogue: bias + relu, D layout M = v + 8*hi
#pragma unroll
    for (int nt = 0; nt < 4; ++nt) {
        int col = nhalf*64 + nt*16 + lrow;
        int p   = base + col;
#pragma unroll
        for (int v = 0; v < 8; ++v) {
            int ch = mtile*16 + hi*8 + v;
            float val = acc[nt][v] + fb[ch];
            xf[(size_t)ch * NPOS1 + p] = val > 0.f ? val : 0.f;
        }
    }
}

// =======================================================================
// Kernel 2: cls conv as GEMM [32 x 576] x [576 x 128-pos tile], fused
//           channel-max; double-buffered B panel, one barrier per step
// =======================================================================
__global__ __launch_bounds__(256) void k_cls_conv(
    const float* __restrict__ xf, const float* __restrict__ cw,
    const float* __restrict__ cb, float* __restrict__ cf,
    float* __restrict__ cmax)
{
    __shared__ __bf16 A[NCLS * ASTR2];
    __shared__ __bf16 Bc[2][128 * KSTR];
    __shared__ float  Ct[NCLS][128];

    const int tid   = threadIdx.x;
    const int lane  = tid & 31, wv = tid >> 5;
    const int lrow  = lane & 15, hi = lane >> 4;
    const int mtile = wv & 1,  nq = wv >> 1;
    const int base  = blockIdx.x * 128;

    for (int u = tid; u < NCLS * KDIM2 / 4; u += 256) {
        int m = u / 144, kq = u - m * 144;
        float4 f = *(const float4*)(cw + m * KDIM2 + 4 * kq);
        v4bf c; c[0]=(__bf16)f.x; c[1]=(__bf16)f.y; c[2]=(__bf16)f.z; c[3]=(__bf16)f.w;
        *(v4bf*)&A[m * ASTR2 + 4 * kq] = c;
    }

    stage_cls(xf, &Bc[0][0], 0, base, tid);
    __syncthreads();

    v8f zero = {0.f,0.f,0.f,0.f,0.f,0.f,0.f,0.f};
    v8f acc[2];
    acc[0] = zero; acc[1] = zero;

    for (int ks = 0; ks < 18; ++ks) {
        const int cur = ks & 1;
        if (ks < 17) stage_cls(xf, &Bc[cur ^ 1][0], (ks + 1) * 32, base, tid);

        v16bf a = ldfragA(&A[(mtile*16 + lrow) * ASTR2 + ks * 32], hi);
        const __bf16* bp = &Bc[cur][(nq*32 + lrow) * KSTR];
        v16bf b0 = ldfragB(bp, hi);
        v16bf b1 = ldfragB(bp + 16 * KSTR, hi);
        acc[0] = __builtin_amdgcn_wmma_f32_16x16x32_bf16(false, a, false, b0, (short)0, acc[0], false, false);
        acc[1] = __builtin_amdgcn_wmma_f32_16x16x32_bf16(false, a, false, b1, (short)0, acc[1], false, false);
        if (ks < 17) __syncthreads();
    }

    // spill D tiles to LDS for channel-max reduction
#pragma unroll
    for (int nt = 0; nt < 2; ++nt) {
        int col = nq*32 + nt*16 + lrow;
#pragma unroll
        for (int v = 0; v < 8; ++v)
            Ct[mtile*16 + hi*8 + v][col] = acc[nt][v];
    }
    __syncthreads();

    for (int e = tid; e < NCLS * 128; e += 256) {
        int ch = e >> 7, col = e & 127;
        int p  = base + col;
        if (p < NPOS2) cf[(size_t)ch * NPOS2 + p] = Ct[ch][col] + cb[ch];
    }
    if (tid < 128) {
        int p = base + tid;
        if (p < NPOS2) {
            float m = -3.4e38f;
#pragma unroll
            for (int ch = 0; ch < NCLS; ++ch)
                m = fmaxf(m, Ct[ch][tid] + cb[ch]);
            cmax[p] = m;
        }
    }
}

// =======================================================================
// Kernel 3: top-128 via candidate tree (tie -> smallest index)
// =======================================================================
#define TCH 64
__global__ __launch_bounds__(1024) void k_topk(float* __restrict__ cmax,
                                               int* __restrict__ idx)
{
    __shared__ float rv[1024], rv2[1024];
    __shared__ int   ri[1024], ri2[1024];
    const int t  = threadIdx.x;
    const int lo = t * TCH;
    const int hi = (lo + TCH < NPOS2) ? lo + TCH : NPOS2;

    float bv = -3.4e38f; int bi = 0x7fffffff;
    for (int j = lo; j < hi; ++j) {
        float v = cmax[j];
        if (v > bv) { bv = v; bi = j; }
    }
    rv[t] = bv; ri[t] = bi;
    __syncthreads();

    for (int it = 0; it < TOPK; ++it) {
        rv2[t] = rv[t]; ri2[t] = ri[t];
        __syncthreads();
        for (int s = 512; s > 0; s >>= 1) {
            if (t < s) {
                float v2 = rv2[t + s]; int i2 = ri2[t + s];
                if (v2 > rv2[t] || (v2 == rv2[t] && i2 < ri2[t])) { rv2[t] = v2; ri2[t] = i2; }
            }
            __syncthreads();
        }
        const int w = ri2[0];
        if (t == 0) idx[it] = w;
        if (t == w / TCH) {
            cmax[w] = -3.4e38f;
            float nb = -3.4e38f; int ni = 0x7fffffff;
            for (int j = lo; j < hi; ++j) {
                float v = cmax[j];
                if (v > nb) { nb = v; ni = j; }
            }
            rv[t] = nb; ri[t] = ni;
        }
        __syncthreads();
    }
}

// =======================================================================
// Kernel 4: per-selected-patch L2 norm -> regn[k][f] = patch[f]/||patch||
// =======================================================================
__global__ __launch_bounds__(576) void k_patchnorm(
    const float* __restrict__ xf, const int* __restrict__ idx,
    float* __restrict__ regn)
{
    __shared__ float red[KDIM2];
    const int k = blockIdx.x, f = threadIdx.x;
    const int fi = idx[k];
    const int r0 = fi / OH2, c0 = fi - r0 * OH2;
    const int c = f / 9, r = f - c * 9, fy = r / 3, fx = r - fy * 3;
    const float val = xf[(size_t)c * NPOS1 + (r0 + fy) * OH + (c0 + fx)];
    red[f] = val * val;
    __syncthreads();
    for (int s = 512; s > 0; s >>= 1) {
        if (f < s && f + s < KDIM2) red[f] += red[f + s];
        __syncthreads();
    }
    const float nrm = fmaxf(sqrtf(red[0]), 1e-12f);
    regn[k * KDIM2 + f] = val / nrm;
}

// =======================================================================
// Kernel 5: cls_out gather -> d_out[2048 + ch*128 + k]
// =======================================================================
__global__ __launch_bounds__(256) void k_clsgather(
    const float* __restrict__ cf, const int* __restrict__ idx,
    float* __restrict__ out)
{
    const int e  = blockIdx.x * 256 + threadIdx.x;
    const int ch = e >> 7, k = e & 127;
    out[FDIM + e] = cf[(size_t)ch * NPOS2 + idx[k]];
}

// =======================================================================
// Kernel 6: t[f] = sum_k regn[k][f] + 128*shift[f]  (fixed-order sum)
// =======================================================================
__global__ __launch_bounds__(576) void k_sumreg(
    const float* __restrict__ regn, const float* __restrict__ shift,
    float* __restrict__ tvec)
{
    const int f = threadIdx.x;
    float s = 0.f;
    for (int k = 0; k < TOPK; ++k) s += regn[k * KDIM2 + f];
    tvec[f] = s + 128.f * shift[f];
}

// =======================================================================
// Kernel 7: acc[j] = lin_w[j,:] . t + 128*lin_b[j]   (t cached in LDS)
// =======================================================================
__global__ __launch_bounds__(256) void k_matvec(
    const float* __restrict__ lw, const float* __restrict__ lb,
    const float* __restrict__ tvec, float* __restrict__ accv)
{
    __shared__ float tl[KDIM2];
    const int t = threadIdx.x;
    for (int f = t; f < KDIM2; f += 256) tl[f] = tvec[f];
    __syncthreads();

    const int j = blockIdx.x * 256 + t;
    const float4* row = (const float4*)(lw + (size_t)j * KDIM2);
    float s = 0.f;
#pragma unroll 4
    for (int q = 0; q < KDIM2 / 4; ++q) {
        float4 w4 = row[q];
        s = fmaf(w4.x, tl[4*q+0], s);
        s = fmaf(w4.y, tl[4*q+1], s);
        s = fmaf(w4.z, tl[4*q+2], s);
        s = fmaf(w4.w, tl[4*q+3], s);
    }
    accv[j] = s + 128.f * lb[j];
}

// =======================================================================
// Kernel 8: desc = acc / max(||acc||, eps)  (deterministic tree reduce)
// =======================================================================
__global__ __launch_bounds__(256) void k_finalnorm(
    const float* __restrict__ accv, float* __restrict__ out)
{
    __shared__ float red[256];
    const int t = threadIdx.x;
    float ss = 0.f;
#pragma unroll
    for (int i = 0; i < 8; ++i) {
        float v = accv[t + i * 256];
        ss = fmaf(v, v, ss);
    }
    red[t] = ss;
    __syncthreads();
    for (int s = 128; s > 0; s >>= 1) {
        if (t < s) red[t] += red[t + s];
        __syncthreads();
    }
    const float denom = fmaxf(sqrtf(red[0]), 1e-12f);
#pragma unroll
    for (int i = 0; i < 8; ++i) {
        int j = t + i * 256;
        out[j] = accv[j] / denom;
    }
}

// =======================================================================
extern "C" void kernel_launch(void* const* d_in, const int* in_sizes, int n_in,
                              void* d_out, int out_size, void* d_ws, size_t ws_size,
                              hipStream_t stream)
{
    (void)in_sizes; (void)n_in; (void)out_size; (void)ws_size;
    const float* x1    = (const float*)d_in[0];
    const float* fw    = (const float*)d_in[1];
    const float* fb    = (const float*)d_in[2];
    const float* cw    = (const float*)d_in[3];
    const float* cb    = (const float*)d_in[4];
    const float* shift = (const float*)d_in[5];
    const float* lw    = (const float*)d_in[6];
    const float* lb    = (const float*)d_in[7];
    float* out = (float*)d_out;

    float* ws   = (float*)d_ws;
    float* xf   = ws + WS_X;
    float* cf   = ws + WS_C;
    float* cmax = ws + WS_CMAX;
    int*   idx  = (int*)(ws + WS_IDX);
    float* regn = ws + WS_REGN;
    float* tvec = ws + WS_T;
    float* accv = ws + WS_ACC;

    k_feat_conv <<<dim3(NPOS1 / 128), dim3(256), 0, stream>>>(x1, fw, fb, xf);
    k_cls_conv  <<<dim3((NPOS2 + 127) / 128), dim3(256), 0, stream>>>(xf, cw, cb, cf, cmax);
    k_topk      <<<dim3(1),   dim3(1024), 0, stream>>>(cmax, idx);
    k_patchnorm <<<dim3(TOPK), dim3(KDIM2), 0, stream>>>(xf, idx, regn);
    k_clsgather <<<dim3(16),  dim3(256), 0, stream>>>(cf, idx, out);
    k_sumreg    <<<dim3(1),   dim3(KDIM2), 0, stream>>>(regn, shift, tvec);
    k_matvec    <<<dim3(FDIM / 256), dim3(256), 0, stream>>>(lw, lb, tvec, accv);
    k_finalnorm <<<dim3(1),   dim3(256), 0, stream>>>(accv, out);
}
